// VecKM_14791867367790
// MI455X (gfx1250) — matrix operand order repaired and checked by reading, not compiled, not run on hardware
//
#include <hip/hip_runtime.h>

// ---------------------------------------------------------------------------
// VecKM chain on gfx1250: T = eB^T eA ; G = eB T ; out = normalize(G / eA)
// eA/eB tiles regenerated on the fly (pts,A,B tiny & L2-resident); GEMMs on
// v_wmma_f32_16x16x32_f16 with fp32 accumulate. K-step 64, and
// sched_group_barrier hints pipeline "2 ds_load_b128 ahead of each wmma".
// Workspace: Tt (fp16, 512x8192, scaled 1/128) = 8MB ; G (fp32, 16384x512) = 32MB
// ---------------------------------------------------------------------------

typedef __attribute__((ext_vector_type(16))) _Float16 v16h;
typedef __attribute__((ext_vector_type(8)))  float    v8f;

#define N_PTS 16384
#define DCOL  256     // D
#define PCOL  4096    // P
#define D2    512     // 2D
#define P2    8192    // 2P
#define KSTEP 64      // K elements staged per barrier (2 WMMA slices)
#define LDK   80      // padded row length (halfs): 160B rows keep 16B alignment

// Scheduler shaping: 0x100 = DS-read group, 0x008 = matrix (WMMA) group.
#if defined(__has_builtin)
#if __has_builtin(__builtin_amdgcn_sched_group_barrier)
#define SGB(m, s) __builtin_amdgcn_sched_group_barrier((m), (s), 0)
#define HAVE_SGB 1
#endif
#endif
#ifndef HAVE_SGB
#define SGB(m, s)
#endif
// Per K-slice: 18 ds reads (2 afrag + 16 bfrag), 8 wmma.
// Lead 6 reads, then alternate (1 wmma, 2 reads) so loads stay 1 frag ahead.
#define PIPELINE_HINT()            \
  do {                             \
    SGB(0x100, 6);                 \
    SGB(0x008, 1); SGB(0x100, 2);  \
    SGB(0x008, 1); SGB(0x100, 2);  \
    SGB(0x008, 1); SGB(0x100, 2);  \
    SGB(0x008, 1); SGB(0x100, 2);  \
    SGB(0x008, 1); SGB(0x100, 2);  \
    SGB(0x008, 1); SGB(0x100, 2);  \
    SGB(0x008, 1);                 \
    SGB(0x008, 1);                 \
  } while (0)

// ---------------------------------------------------------------------------
// GEMM1: T[p][d] = sum_n eB[n][p] * eA[n][d]   (M=8192, N=512, K=16384)
// Writes Tt[d][p] = T[p][d] / 128 as fp16 (transposed for GEMM2 B-side).
// ---------------------------------------------------------------------------
__global__ __launch_bounds__(256)
void vkm_gemm1(const float* __restrict__ pts, const float* __restrict__ Amat,
               const float* __restrict__ Bmat, _Float16* __restrict__ Tt) {
  __shared__ __align__(16) _Float16 lA[128][LDK]; // eB^T tile: [p_local][k=n]
  __shared__ __align__(16) _Float16 lB[128][LDK]; // eA   tile: [d_local][k=n]

  const int tid  = threadIdx.x;
  const int lane = tid & 31;
  const int wave = tid >> 5;            // 0..7
  const int p0   = blockIdx.x * 128;    // row tile of T  (p)
  const int d0   = blockIdx.y * 128;    // col tile of T  (d)

  // ---- tile-generation assignment: thread -> (row gj, k-chunk gk) ----
  const int gj = tid >> 1;              // 0..127
  const int gk = (tid & 1) * 32;        // 0 or 32

  const int   dg   = d0 + gj;
  const int   dcol = dg & (DCOL - 1);
  const float a0 = Amat[0 * DCOL + dcol];
  const float a1 = Amat[1 * DCOL + dcol];
  const float a2 = Amat[2 * DCOL + dcol];
  const bool  dcos = dg < DCOL;

  const int   pg   = p0 + gj;
  const int   pcol = pg & (PCOL - 1);
  const float b0 = Bmat[0 * PCOL + pcol];
  const float b1 = Bmat[1 * PCOL + pcol];
  const float b2 = Bmat[2 * PCOL + pcol];
  const bool  pcos = pg < PCOL;

  v8f acc[8] = {};

  // fragment addressing (ISA 16-bit A 16x32 / B 32x16 lane layouts)
  const int prow   = wave * 16 + (lane & 15);
  const int kbaseA = (lane < 16) ? 0 : 8;
  const int kbaseB = (lane >> 4) * 8;

  for (int kt = 0; kt < N_PTS / KSTEP; ++kt) {
    const int n0 = kt * KSTEP;
    __syncthreads();
    // regenerate eB^T and eA tiles for n in [n0, n0+64)
#pragma unroll 8
    for (int i = 0; i < 32; ++i) {
      const int   kk = gk + i;
      const float x0 = pts[(n0 + kk) * 3 + 0];
      const float x1 = pts[(n0 + kk) * 3 + 1];
      const float x2 = pts[(n0 + kk) * 3 + 2];
      const float xa = x0 * a0 + x1 * a1 + x2 * a2;
      const float xb = x0 * b0 + x1 * b1 + x2 * b2;
      lB[gj][kk] = (_Float16)(dcos ? __cosf(xa) : __sinf(xa));
      lA[gj][kk] = (_Float16)(pcos ? __cosf(xb) : __sinf(xb));
    }
    __syncthreads();

#pragma unroll
    for (int kh = 0; kh < 2; ++kh) {
      const int kb = kh * 32;
      v16h afrag;
      *((uint4*)&afrag)       = *((const uint4*)&lA[prow][kb + kbaseA]);
      *(((uint4*)&afrag) + 1) = *((const uint4*)&lA[prow][kb + kbaseA + 16]);
      v16h bf[8];
#pragma unroll
      for (int g = 0; g < 8; ++g) {
        const int drow = g * 16 + (lane & 15);
        *((uint4*)&bf[g])       = *((const uint4*)&lB[drow][kb + kbaseB]);
        *(((uint4*)&bf[g]) + 1) = *((const uint4*)&lB[drow][kb + kbaseB + 16]);
      }
#pragma unroll
      for (int g = 0; g < 8; ++g)
        acc[g] = __builtin_amdgcn_wmma_f32_16x16x32_f16(
            false, afrag, false, bf[g], (short)0, acc[g], false, false);
      PIPELINE_HINT();
    }
  }

  // epilogue: Tt[d][p] = acc/128 (fp16)
  const int mrow = (lane >> 4) * 8;
#pragma unroll
  for (int g = 0; g < 8; ++g) {
    const int d = d0 + g * 16 + (lane & 15);
#pragma unroll
    for (int r = 0; r < 8; ++r) {
      const int p = p0 + wave * 16 + mrow + r;
      Tt[(size_t)d * P2 + p] = (_Float16)(acc[g][r] * 0.0078125f);
    }
  }
}

// ---------------------------------------------------------------------------
// GEMM2: G[n][d] = sum_p eB[n][p] * T[p][d]   (M=16384, N=512, K=8192)
// A-side eB regenerated on the fly; B-side T read from Tt (fp16, [d][p]).
// ---------------------------------------------------------------------------
__global__ __launch_bounds__(256)
void vkm_gemm2(const float* __restrict__ pts, const float* __restrict__ Bmat,
               const _Float16* __restrict__ Tt, float* __restrict__ G) {
  __shared__ __align__(16) _Float16 lA[128][LDK]; // eB tile: [n_local][k=p]
  __shared__ __align__(16) _Float16 lB[128][LDK]; // T  tile: [d_local][k=p]

  const int tid  = threadIdx.x;
  const int lane = tid & 31;
  const int wave = tid >> 5;
  const int n0   = blockIdx.x * 128;    // row tile (n)
  const int d0   = blockIdx.y * 128;    // col tile (d)

  const int gj = tid >> 1;
  const int gk = (tid & 1) * 32;

  // this thread generates eB for one fixed point n0+gj
  const int   n  = n0 + gj;
  const float x0 = pts[n * 3 + 0];
  const float x1 = pts[n * 3 + 1];
  const float x2 = pts[n * 3 + 2];

  v8f acc[8] = {};

  const int nrow   = wave * 16 + (lane & 15);
  const int kbaseA = (lane < 16) ? 0 : 8;
  const int kbaseB = (lane >> 4) * 8;

  for (int kt = 0; kt < P2 / KSTEP; ++kt) {
    const int pk0 = kt * KSTEP;
    __syncthreads();
    // B-side: copy Tt[d0+gj][pk0+gk .. +31] into LDS (64B contiguous)
    {
      const _Float16* src = &Tt[(size_t)(d0 + gj) * P2 + pk0 + gk];
#pragma unroll
      for (int q = 0; q < 4; ++q)
        *((uint4*)&lB[gj][gk + 8 * q]) = *((const uint4*)(src + 8 * q));
    }
    // A-side: regenerate eB[n][pk0+gk .. +31] (chunk never crosses cos/sin split)
    {
      const int  pbase = pk0 + gk;
      const bool pcos  = pbase < PCOL;
#pragma unroll 8
      for (int i = 0; i < 32; ++i) {
        const int   pc = (pbase + i) & (PCOL - 1);
        const float c0 = Bmat[0 * PCOL + pc];
        const float c1 = Bmat[1 * PCOL + pc];
        const float c2 = Bmat[2 * PCOL + pc];
        const float xb = x0 * c0 + x1 * c1 + x2 * c2;
        lA[gj][gk + i] = (_Float16)(pcos ? __cosf(xb) : __sinf(xb));
      }
    }
    __syncthreads();

#pragma unroll
    for (int kh = 0; kh < 2; ++kh) {
      const int kb = kh * 32;
      v16h afrag;
      *((uint4*)&afrag)       = *((const uint4*)&lA[nrow][kb + kbaseA]);
      *(((uint4*)&afrag) + 1) = *((const uint4*)&lA[nrow][kb + kbaseA + 16]);
      v16h bf[8];
#pragma unroll
      for (int g = 0; g < 8; ++g) {
        const int drow = g * 16 + (lane & 15);
        *((uint4*)&bf[g])       = *((const uint4*)&lB[drow][kb + kbaseB]);
        *(((uint4*)&bf[g]) + 1) = *((const uint4*)&lB[drow][kb + kbaseB + 16]);
      }
#pragma unroll
      for (int g = 0; g < 8; ++g)
        acc[g] = __builtin_amdgcn_wmma_f32_16x16x32_f16(
            false, afrag, false, bf[g], (short)0, acc[g], false, false);
      PIPELINE_HINT();
    }
  }

  // epilogue: G[n][d] fp32 (scale irrelevant: normalized later)
  const int mrow = (lane >> 4) * 8;
#pragma unroll
  for (int g = 0; g < 8; ++g) {
    const int d = d0 + g * 16 + (lane & 15);
#pragma unroll
    for (int r = 0; r < 8; ++r) {
      const int nn = n0 + wave * 16 + mrow + r;
      G[(size_t)nn * D2 + d] = acc[g][r];
    }
  }
}

// ---------------------------------------------------------------------------
// Finalize: Gc = (Gr + i Gi) * (cos - i sin)   (|eA element| == 1, so the
// complex division is a conjugate rotation and preserves the norm),
// then scale row to norm sqrt(D)=16. One wave per point.
// ---------------------------------------------------------------------------
__global__ __launch_bounds__(256)
void vkm_finalize(const float* __restrict__ pts, const float* __restrict__ Amat,
                  const float* __restrict__ G, float* __restrict__ out) {
  const int lane = threadIdx.x & 31;
  const int wave = threadIdx.x >> 5;
  const int n    = blockIdx.x * 8 + wave;

  const float x0 = pts[n * 3 + 0];
  const float x1 = pts[n * 3 + 1];
  const float x2 = pts[n * 3 + 2];

  float re[8], im[8];
  float nrm = 0.f;
#pragma unroll
  for (int i = 0; i < 8; ++i) {
    const int   d  = lane + i * 32;
    const float gr = G[(size_t)n * D2 + d];
    const float gi = G[(size_t)n * D2 + DCOL + d];
    const float xa = x0 * Amat[d] + x1 * Amat[DCOL + d] + x2 * Amat[2 * DCOL + d];
    const float c  = __cosf(xa);
    const float s  = __sinf(xa);
    re[i] = gr * c + gi * s;
    im[i] = gi * c - gr * s;
    nrm  += gr * gr + gi * gi;       // == |Gc|^2 (unit-modulus rotation)
  }
  for (int off = 16; off > 0; off >>= 1)
    nrm += __shfl_xor(nrm, off, 32);
  const float scale = 16.0f * rsqrtf(nrm);
#pragma unroll
  for (int i = 0; i < 8; ++i) {
    const int d = lane + i * 32;
    out[(size_t)n * D2 + 2 * d]     = re[i] * scale;  // complex64 interleaved
    out[(size_t)n * D2 + 2 * d + 1] = im[i] * scale;
  }
}

// ---------------------------------------------------------------------------
extern "C" void kernel_launch(void* const* d_in, const int* in_sizes, int n_in,
                              void* d_out, int out_size, void* d_ws, size_t ws_size,
                              hipStream_t stream) {
  const float* pts  = (const float*)d_in[0];   // (1,16384,3) f32
  const float* Amat = (const float*)d_in[1];   // (3,256)     f32
  const float* Bmat = (const float*)d_in[2];   // (3,4096)    f32
  float*       out  = (float*)d_out;           // 16384x256 complex64 interleaved

  _Float16* Tt = (_Float16*)d_ws;                                   // 8 MB
  float*    G  = (float*)((char*)d_ws + (size_t)D2 * P2 * 2);       // 32 MB

  dim3 g1(P2 / 128, D2 / 128);      // 64 x 4
  vkm_gemm1<<<g1, 256, 0, stream>>>(pts, Amat, Bmat, Tt);

  dim3 g2(N_PTS / 128, D2 / 128);   // 128 x 4
  vkm_gemm2<<<g2, 256, 0, stream>>>(pts, Bmat, Tt, G);

  vkm_finalize<<<N_PTS / 8, 256, 0, stream>>>(pts, Amat, G, out);
}
